// SLAAttentionImpl_61632780697903
// MI455X (gfx1250) — matrix operand order repaired and checked by reading, not compiled, not run on hardware
//
#include <hip/hip_runtime.h>
#include <hip/hip_bf16.h>
#include <math.h>

#define HNUM 24
#define LLEN 2048
#define DDIM 128
#define NBLK 32
#define BLK  64
#define TOPK 16
#define SCALE 0.08838834764831845f   /* 1/sqrt(128) */

typedef __attribute__((ext_vector_type(16))) __bf16 v16bf;
typedef __attribute__((ext_vector_type(8)))  float  v8f;

union FragBF { v16bf v; unsigned short h[16]; unsigned int u[8]; };

__device__ __forceinline__ v8f zero8() { v8f z = {}; return z; }

__device__ __forceinline__ unsigned short f2bf(float f) {
  unsigned int u = __float_as_uint(f);
  u += 0x7FFFu + ((u >> 16) & 1u);           // round-to-nearest-even
  return (unsigned short)(u >> 16);
}
__device__ __forceinline__ float bf2f(unsigned short b) {
  return __uint_as_float((unsigned int)b << 16);
}

// A-matrix (16x32 bf16) fragment from row-major bf16 storage.
// ISA layout: lane<16 -> K in {0..7} U {16..23}; lane>=16 -> K in {8..15} U {24..31}; row M = lane%16.
__device__ __forceinline__ v16bf frag_a_lds(const unsigned short* base, int ld, int row0, int k0) {
  const int lane = threadIdx.x & 31;
  const int m = lane & 15;
  const int kb = (lane & 16) ? 8 : 0;
  const unsigned short* p = base + (row0 + m) * ld + k0;
  FragBF f;
#pragma unroll
  for (int j = 0; j < 4; ++j) {
    f.u[j]     = *(const unsigned int*)(p + kb + 2 * j);
    f.u[j + 4] = *(const unsigned int*)(p + 16 + kb + 2 * j);
  }
  return f.v;
}

// Same A fragment, loaded from row-major f32 global memory with on-the-fly bf16 convert.
__device__ __forceinline__ v16bf frag_a_f32(const float* base, int ld, int row0, int k0) {
  const int lane = threadIdx.x & 31;
  const int m = lane & 15;
  const int kb = (lane & 16) ? 8 : 0;
  const float* p = base + (size_t)(row0 + m) * ld + k0;
  FragBF f;
#pragma unroll
  for (int j = 0; j < 8; ++j) {
    int kk = (j < 4) ? (kb + 2 * j) : (16 + kb + 2 * (j - 4));
    f.h[2 * j]     = f2bf(p[kk]);
    f.h[2 * j + 1] = f2bf(p[kk + 1]);
  }
  return f.v;
}

// B-matrix (32x16 bf16) fragment from N-major (transposed) bf16 storage Bt[n][k].
// Layout: lane<16 -> K {0..7} U {16..23}, lane>=16 -> K {8..15} U {24..31}; column N = lane%16; K packed in pairs.
__device__ __forceinline__ v16bf frag_b_t(const unsigned short* baseT, int ld, int col0, int k0) {
  const int lane = threadIdx.x & 31;
  const int n = lane & 15;
  const int kb = (lane & 16) ? 8 : 0;
  const unsigned short* p = baseT + (col0 + n) * ld + k0 + kb;
  FragBF f;
#pragma unroll
  for (int j = 0; j < 4; ++j) {
    f.u[j]     = *(const unsigned int*)(p + 2 * j);
    f.u[j + 4] = *(const unsigned int*)(p + 16 + 2 * j);
  }
  return f.v;
}

#define WMMA_BF16(a, b, c) \
  __builtin_amdgcn_wmma_f32_16x16x32_bf16(false, (a), false, (b), (short)0, (c), false, false)

// ---------------------------------------------------------------------------
// Kernel 1: block-mean pooling, 32x32 block score map, deterministic top-16 LUT
// ---------------------------------------------------------------------------
__global__ void __launch_bounds__(256) slaa_pool_topk(const float* __restrict__ q,
                                                      const float* __restrict__ k,
                                                      int* __restrict__ lut) {
  __shared__ float qbs[NBLK * DDIM];
  __shared__ float kbs[NBLK * DDIM];
  __shared__ float sc[NBLK * NBLK];
  const int h = blockIdx.x;
  const int t = threadIdx.x;

  for (int idx = t; idx < NBLK * DDIM; idx += 256) {
    int blk = idx >> 7, d = idx & 127;
    float sq = 0.f, sk = 0.f;
    for (int r = 0; r < BLK; ++r) {
      size_t off = ((size_t)(blk * BLK + r) * HNUM + h) * DDIM + d;
      sq += q[off];
      sk += k[off];
    }
    qbs[idx] = sq * (1.0f / BLK);
    kbs[idx] = sk * (1.0f / BLK);
  }
  __syncthreads();
  for (int idx = t; idx < NBLK * NBLK; idx += 256) {
    int qi = idx >> 5, ki = idx & 31;
    float s = 0.f;
    for (int d = 0; d < DDIM; ++d) s += qbs[qi * DDIM + d] * kbs[ki * DDIM + d];
    sc[idx] = s;
  }
  __syncthreads();
  if (t < NBLK) {
    const float* row = &sc[t * NBLK];
    for (int i = 0; i < NBLK; ++i) {
      float si = row[i];
      int rank = 0;
      for (int j = 0; j < NBLK; ++j) {
        float sj = row[j];
        if (sj > si || (sj == si && j < i)) rank++;
      }
      if (rank < TOPK) lut[((h * NBLK) + t) * TOPK + rank] = i;
    }
  }
}

// ---------------------------------------------------------------------------
// Kernel 2: block-sparse flash attention.  grid = (Nq, H), 128 threads (4 waves)
// Each wave owns 16 q-rows; Q A-fragments live in registers for the whole kernel.
// ---------------------------------------------------------------------------
__global__ void __launch_bounds__(128) slaa_sparse(const float* __restrict__ q,
                                                   const float* __restrict__ k,
                                                   const float* __restrict__ v,
                                                   const int* __restrict__ lut,
                                                   float* __restrict__ out) {
  __shared__ __align__(16) unsigned short Ks[BLK * DDIM];   // [key][d]   bf16, row-major
  __shared__ __align__(16) unsigned short VTs[DDIM * BLK];  // [d][key]   bf16, N-major for B
  __shared__ __align__(16) float          Ss[BLK * BLK];    // [qrow][key] f32 scores
  __shared__ __align__(16) unsigned short Ps[BLK * BLK];    // [qrow][key] bf16 probs
  __shared__ float mrow[BLK], lrow[BLK], arow[BLK];

  const int qb   = blockIdx.x;
  const int h    = blockIdx.y;
  const int tid  = threadIdx.x;
  const int wave = tid >> 5;
  const int lane = tid & 31;
  const int crow = (lane >> 4) << 3;   // C-layout: row = r + 8*(lane/16)
  const int ccol = lane & 15;          //           col = lane%16

  // Q fragments (A) held in registers: 4 K-chunks of this wave's 16 rows
  const float* qbase = q + ((size_t)(qb * BLK + wave * 16) * HNUM + h) * DDIM;
  v16bf qf[4];
#pragma unroll
  for (int dc = 0; dc < 4; ++dc) qf[dc] = frag_a_f32(qbase, HNUM * DDIM, 0, dc * 32);

  v8f acc[8];
#pragma unroll
  for (int ct = 0; ct < 8; ++ct) acc[ct] = zero8();
  if (tid < BLK) { mrow[tid] = -INFINITY; lrow[tid] = 0.f; }
  __syncthreads();

  for (int t = 0; t < TOPK; ++t) {
    const int kb = lut[((h * NBLK) + qb) * TOPK + t];

    // stage K row-major and V transposed (f32 -> bf16)
    for (int idx = tid; idx < BLK * DDIM; idx += 128) {
      int r = idx >> 7, d = idx & 127;
      size_t off = ((size_t)(kb * BLK + r) * HNUM + h) * DDIM + d;
      Ks[r * DDIM + d] = f2bf(k[off]);
      VTs[d * BLK + r] = f2bf(v[off]);
    }
    __syncthreads();

    // S = Q K^T : 4 key tiles x 4 K-chunks = 16 WMMAs per wave
#pragma unroll
    for (int kt = 0; kt < 4; ++kt) {
      v8f c = zero8();
#pragma unroll
      for (int dc = 0; dc < 4; ++dc) {
        v16bf bf = frag_b_t(Ks, DDIM, kt * 16, dc * 32);
        c = WMMA_BF16(qf[dc], bf, c);
      }
#pragma unroll
      for (int r = 0; r < 8; ++r)
        Ss[(wave * 16 + crow + r) * BLK + kt * 16 + ccol] = c[r];
    }
    __syncthreads();

    // online softmax: one thread per q-row
    if (tid < BLK) {
      const float* srow = &Ss[tid * BLK];
      unsigned short* prow = &Ps[tid * BLK];
      float mx = -INFINITY;
#pragma unroll 8
      for (int c = 0; c < BLK; ++c) mx = fmaxf(mx, srow[c] * SCALE);
      float mold = mrow[tid];
      float mnew = fmaxf(mold, mx);
      float alpha = __expf(mold - mnew);
      float sum = 0.f;
#pragma unroll 8
      for (int c = 0; c < BLK; ++c) {
        float p = __expf(srow[c] * SCALE - mnew);
        sum += p;
        prow[c] = f2bf(p);
      }
      mrow[tid] = mnew;
      lrow[tid] = lrow[tid] * alpha + sum;
      arow[tid] = alpha;
    }
    __syncthreads();

    // rescale accumulators by exp(m_old - m_new)
    float al[8];
#pragma unroll
    for (int r = 0; r < 8; ++r) al[r] = arow[wave * 16 + crow + r];
#pragma unroll
    for (int ct = 0; ct < 8; ++ct)
#pragma unroll
      for (int r = 0; r < 8; ++r) acc[ct][r] *= al[r];

    // O += P V : A = P (16x64), B = V^T staged N-major; 16 WMMAs per wave
    const unsigned short* pbase = &Ps[wave * 16 * BLK];
    v16bf pf[2];
#pragma unroll
    for (int kc = 0; kc < 2; ++kc) pf[kc] = frag_a_lds(pbase, BLK, 0, kc * 32);
#pragma unroll
    for (int ct = 0; ct < 8; ++ct)
#pragma unroll
      for (int kc = 0; kc < 2; ++kc) {
        v16bf bf = frag_b_t(VTs, BLK, ct * 16, kc * 32);
        acc[ct] = WMMA_BF16(pf[kc], bf, acc[ct]);
      }
    __syncthreads();
  }

  // normalize by row sums and write sparse result (overwrite d_out)
  float li[8];
#pragma unroll
  for (int r = 0; r < 8; ++r) li[r] = 1.0f / lrow[wave * 16 + crow + r];
#pragma unroll
  for (int ct = 0; ct < 8; ++ct)
#pragma unroll
    for (int r = 0; r < 8; ++r) {
      int row = qb * BLK + wave * 16 + crow + r;
      out[((size_t)row * HNUM + h) * DDIM + ct * 16 + ccol] = acc[ct][r] * li[r];
    }
}

// ---------------------------------------------------------------------------
// Kernel 3: linear attention kvsum^T = V^T @ softmax(K) + ksum.  grid = H, 256 thr
// 8 waves each own a 16-row strip of the 128x128 accumulator (8 tiles -> 64 VGPRs).
// ---------------------------------------------------------------------------
__global__ void __launch_bounds__(256) slaa_linkv(const float* __restrict__ k,
                                                  const float* __restrict__ v,
                                                  unsigned short* __restrict__ kvT,
                                                  float* __restrict__ ksum) {
  __shared__ __align__(16) unsigned short klT[DDIM * 32];  // [d][l] bf16 (B, N-major)
  __shared__ __align__(16) unsigned short vT[DDIM * 32];   // [e][l] bf16 (A, row-major)
  __shared__ float ksl[DDIM];
  const int h    = blockIdx.x;
  const int tid  = threadIdx.x;
  const int wave = tid >> 5;
  const int lane = tid & 31;
  const int crow = (lane >> 4) << 3;
  const int ccol = lane & 15;

  if (tid < DDIM) ksl[tid] = 0.f;
  v8f acc[8];
#pragma unroll
  for (int dt = 0; dt < 8; ++dt) acc[dt] = zero8();
  __syncthreads();

  for (int ch = 0; ch < LLEN / 32; ++ch) {
    const int l0 = ch * 32;
    if (tid < 32) {                    // softmax over D for 32 K rows
      const float* krow = k + ((size_t)(l0 + tid) * HNUM + h) * DDIM;
      float mx = -INFINITY;
      for (int d = 0; d < DDIM; ++d) mx = fmaxf(mx, krow[d]);
      float sum = 0.f;
      for (int d = 0; d < DDIM; ++d) sum += __expf(krow[d] - mx);
      float inv = 1.0f / sum;
      for (int d = 0; d < DDIM; ++d)
        klT[d * 32 + tid] = f2bf(__expf(krow[d] - mx) * inv);
    }
    for (int idx = tid; idx < 32 * DDIM; idx += 256) {   // V transpose stage
      int li = idx >> 7, d = idx & 127;
      vT[d * 32 + li] = f2bf(v[((size_t)(l0 + li) * HNUM + h) * DDIM + d]);
    }
    __syncthreads();

    if (tid < DDIM) {                  // ksum accumulation
      float s = 0.f;
      for (int j = 0; j < 32; ++j) s += bf2f(klT[tid * 32 + j]);
      ksl[tid] += s;
    }
    // kvT[e][d] += vT-chunk @ klT-chunk : 8 WMMAs per wave per chunk
    v16bf af = frag_a_lds(vT, 32, wave * 16, 0);
#pragma unroll
    for (int dt = 0; dt < 8; ++dt) {
      v16bf bf = frag_b_t(klT, 32, dt * 16, 0);
      acc[dt] = WMMA_BF16(af, bf, acc[dt]);
    }
    __syncthreads();
  }

#pragma unroll
  for (int dt = 0; dt < 8; ++dt)
#pragma unroll
    for (int r = 0; r < 8; ++r) {
      int e = wave * 16 + crow + r;
      int d = dt * 16 + ccol;
      kvT[((size_t)h * DDIM + e) * DDIM + d] = f2bf(acc[dt][r]);
    }
  if (tid < DDIM) ksum[h * DDIM + tid] = ksl[tid];
}

// ---------------------------------------------------------------------------
// Kernel 4: num = softmax(Q) @ kvsum, normalize by den, project by W^T, add bias,
// accumulate into d_out (already holds sparse result).  grid = (Nq, H), 128 thr.
// ---------------------------------------------------------------------------
__global__ void __launch_bounds__(128) slaa_linout(const float* __restrict__ q,
                                                   const unsigned short* __restrict__ kvT,
                                                   const float* __restrict__ ksum,
                                                   const float* __restrict__ pw,
                                                   const float* __restrict__ pb,
                                                   float* __restrict__ out) {
  __shared__ __align__(16) unsigned short qls[BLK * DDIM];   // softmax(Q) bf16, then o_l bf16
  __shared__ __align__(16) unsigned short Wls[DDIM * DDIM];  // proj_w [e][d] bf16 (N-major B)
  __shared__ float den[BLK];
  const int cb   = blockIdx.x;
  const int h    = blockIdx.y;
  const int tid  = threadIdx.x;
  const int wave = tid >> 5;
  const int lane = tid & 31;
  const int crow = (lane >> 4) << 3;
  const int ccol = lane & 15;

  for (int idx = tid; idx < DDIM * DDIM; idx += 128) Wls[idx] = f2bf(pw[idx]);

  if (tid < BLK) {                     // softmax(Q) row + denominator
    const float* qrow = q + ((size_t)(cb * BLK + tid) * HNUM + h) * DDIM;
    float mx = -INFINITY;
    for (int d = 0; d < DDIM; ++d) mx = fmaxf(mx, qrow[d]);
    float sum = 0.f;
    for (int d = 0; d < DDIM; ++d) sum += __expf(qrow[d] - mx);
    float inv = 1.0f / sum;
    float dn = 0.f;
    for (int d = 0; d < DDIM; ++d) {
      float e = __expf(qrow[d] - mx) * inv;
      qls[tid * DDIM + d] = f2bf(e);
      dn += e * ksum[h * DDIM + d];
    }
    den[tid] = 1e-5f + dn;
  }
  __syncthreads();

  // GEMM1: num = ql @ kvsum  (B = kvT[e][d] bf16 from workspace)
  v8f acc[8];
#pragma unroll
  for (int ct = 0; ct < 8; ++ct) acc[ct] = zero8();
  v16bf af[4];
#pragma unroll
  for (int dc = 0; dc < 4; ++dc) af[dc] = frag_a_lds(qls, DDIM, wave * 16, dc * 32);
  const unsigned short* kvh = kvT + (size_t)h * DDIM * DDIM;
#pragma unroll
  for (int ct = 0; ct < 8; ++ct)
#pragma unroll
    for (int dc = 0; dc < 4; ++dc) {
      v16bf bf = frag_b_t(kvh, DDIM, ct * 16, dc * 32);
      acc[ct] = WMMA_BF16(af[dc], bf, acc[ct]);
    }
  __syncthreads();   // all waves done reading qls

  // normalize by den, restage o_l as bf16 A matrix
  float dinv[8];
#pragma unroll
  for (int r = 0; r < 8; ++r) dinv[r] = 1.0f / den[wave * 16 + crow + r];
#pragma unroll
  for (int ct = 0; ct < 8; ++ct)
#pragma unroll
    for (int r = 0; r < 8; ++r)
      qls[(wave * 16 + crow + r) * DDIM + ct * 16 + ccol] = f2bf(acc[ct][r] * dinv[r]);
  __syncthreads();

  // GEMM2: o_l @ W^T   (B = Wls[e][d])
  v8f acc2[8];
#pragma unroll
  for (int ct = 0; ct < 8; ++ct) acc2[ct] = zero8();
#pragma unroll
  for (int dc = 0; dc < 4; ++dc) af[dc] = frag_a_lds(qls, DDIM, wave * 16, dc * 32);
#pragma unroll
  for (int ct = 0; ct < 8; ++ct)
#pragma unroll
    for (int dc = 0; dc < 4; ++dc) {
      v16bf bf = frag_b_t(Wls, DDIM, ct * 16, dc * 32);
      acc2[ct] = WMMA_BF16(af[dc], bf, acc2[ct]);
    }

  // add linear branch (+ bias) into output
#pragma unroll
  for (int ct = 0; ct < 8; ++ct)
#pragma unroll
    for (int r = 0; r < 8; ++r) {
      int row = cb * BLK + wave * 16 + crow + r;
      int e = ct * 16 + ccol;
      size_t off = ((size_t)row * HNUM + h) * DDIM + e;
      out[off] = out[off] + acc2[ct][r] + pb[e];
    }
}

// ---------------------------------------------------------------------------
extern "C" void kernel_launch(void* const* d_in, const int* in_sizes, int n_in,
                              void* d_out, int out_size, void* d_ws, size_t ws_size,
                              hipStream_t stream) {
  (void)in_sizes; (void)n_in; (void)out_size; (void)ws_size;
  const float* q  = (const float*)d_in[0];
  const float* k  = (const float*)d_in[1];
  const float* v  = (const float*)d_in[2];
  const float* pw = (const float*)d_in[3];
  const float* pb = (const float*)d_in[4];
  float* out = (float*)d_out;

  char* ws = (char*)d_ws;
  int*            lut  = (int*)ws;                          // 24*32*16*4      = 49152 B
  float*          ksum = (float*)(ws + 49152);              // 24*128*4        = 12288 B
  unsigned short* kvT  = (unsigned short*)(ws + 61440);     // 24*128*128*2    = 786432 B

  slaa_pool_topk<<<dim3(HNUM), dim3(256), 0, stream>>>(q, k, lut);
  slaa_sparse<<<dim3(NBLK, HNUM), dim3(128), 0, stream>>>(q, k, v, lut, out);
  slaa_linkv<<<dim3(HNUM), dim3(256), 0, stream>>>(k, v, kvT, ksum);
  slaa_linout<<<dim3(NBLK, HNUM), dim3(128), 0, stream>>>(q, kvT, ksum, pw, pb, out);
}
